// RNNmodel_39831526703407
// MI455X (gfx1250) — matrix operand order repaired and checked by reading, not compiled, not run on hardware
//
#include <hip/hip_runtime.h>
#include <hip/hip_bf16.h>
#include <math.h>

typedef __attribute__((ext_vector_type(16))) _Float16 v16h;
typedef __attribute__((ext_vector_type(2)))  _Float16 v2h;
typedef __attribute__((ext_vector_type(8)))  float    v8f;

#define VOCAB  27
#define EMBED  48
#define HIDDEN 128
#define TLEN   2048
#define HPAD   136   // padded f16 stride for h in LDS (272B rows -> conflict-free A loads)
#define CHUNK  64    // timesteps of x indices staged in LDS per refill

// Branch-free tanh on the hardware transcendental path.
// Prefer gfx1250's V_TANH_F32; else 1 - 2/(exp2(2*log2e*x)+1) via v_exp_f32/v_rcp_f32
// (saturates to +/-1 through inf/0 without any compares).
static __device__ __forceinline__ float fast_tanh(float v) {
#if __has_builtin(__builtin_amdgcn_tanhf)
  return __builtin_amdgcn_tanhf(v);
#elif __has_builtin(__builtin_amdgcn_tanh_f32)
  return __builtin_amdgcn_tanh_f32(v);
#else
  const float p = __builtin_amdgcn_exp2f(2.88539008177793f * v);  // 2*log2(e)
  return 1.0f - 2.0f * __builtin_amdgcn_rcpf(p + 1.0f);
#endif
}

// One workgroup owns 16 batch rows for all 2048 timesteps.
// 8 waves; wave w owns hidden columns [16w,16w+16). Each step:
//   acc  = lut[x[b,t], cols]                      (input projection via LDS LUT)
//   acc += h_{t-1}[16x128] @ W_hh^T[:,cols]       (4x v_wmma_f32_16x16x32_f16)
//   h_t  = tanh(acc) -> LDS ping-pong buffer -> reload as A-fragments
//   waves 0,1 reuse the fresh A-fragments with W_fc to emit out[:,t,0:27]
__global__ __launch_bounds__(256) void rnn_fused(
    const int* __restrict__ x, const float* __restrict__ embed,
    const float* __restrict__ W_ih, const float* __restrict__ W_hh,
    const float* __restrict__ b_ih, const float* __restrict__ b_hh,
    const float* __restrict__ W_fc, const float* __restrict__ b_fc,
    float* __restrict__ out, float* __restrict__ hT)
{
  __shared__ float    lut[VOCAB * HIDDEN];   // 13824 B: W_ih@embed[v] + b_ih + b_hh
  __shared__ _Float16 hbuf[2][16 * HPAD];    //  8704 B: ping-pong h_t (f16)
  __shared__ int      xs[16 * CHUNK];        //  4096 B: staged indices

  const int tid   = threadIdx.x;
  const int wave  = tid >> 5;
  const int lane  = tid & 31;
  const int lhalf = lane >> 4;
  const int lm    = lane & 15;
  const int b0    = blockIdx.x * 16;
  const int nbase = wave * 16;

  // ---- Build the fused input-projection LUT (kills the 6.4 GFLOP e@W_ih GEMM) ----
  for (int i = tid; i < VOCAB * HIDDEN; i += 256) {
    const int v = i >> 7, hc = i & (HIDDEN - 1);
    float s = b_ih[hc] + b_hh[hc];
    const float* wr = W_ih  + hc * EMBED;
    const float* er = embed + v  * EMBED;
    #pragma unroll 8
    for (int e = 0; e < EMBED; ++e) s = fmaf(wr[e], er[e], s);
    lut[i] = s;
  }

  // ---- Resident B-fragments: W_hh^T slice for this wave's 16 columns ----
  // 16-bit B (32x16): lane%16 = N column, element e = K row (table order).  B[k][n] = W_hh[n][k].
  v16h bhh[4];
  #pragma unroll
  for (int kb = 0; kb < 4; ++kb) {
    const int kk = kb * 32 + 8 * lhalf;
    #pragma unroll
    for (int e = 0; e < 16; ++e)
      bhh[kb][e] = (_Float16)W_hh[(nbase + lm) * HIDDEN + kk + (e & 7) + 16 * (e >> 3)];
  }

  // ---- Head B-fragments (waves 0,1 cover N=0..31, mask >=27) ----
  v16h bfc[4] = {};
  float bias  = 0.0f;
  const int cfc = nbase + lm;
  if (wave < 2) {
    #pragma unroll
    for (int kb = 0; kb < 4; ++kb) {
      const int kk = kb * 32 + 8 * lhalf;
      #pragma unroll
      for (int e = 0; e < 16; ++e) {
        const int n = nbase + lm;
        bfc[kb][e] = (n < VOCAB) ? (_Float16)W_fc[n * HIDDEN + kk + (e & 7) + 16 * (e >> 3)] : (_Float16)0.0f;
      }
    }
    if (cfc < VOCAB) bias = b_fc[cfc];
  }

  // h_0 = 0 as A-fragments in registers
  const v16h zero16 = {};
  v16h a[4];
  #pragma unroll
  for (int kb = 0; kb < 4; ++kb) a[kb] = zero16;

  __syncthreads();

  v8f acc = {};
  const size_t outRowStride = (size_t)TLEN * VOCAB;

  for (int t = 0; t < TLEN; ++t) {
    // ---- Stage next 64 timesteps of indices; prefetch the chunk after ----
    // (Old xs readers are ordered before barrier_{t-1}; one sync after suffices.)
    if ((t & (CHUNK - 1)) == 0) {
      #pragma unroll
      for (int j = 0; j < 4; ++j) {
        const int li = tid + j * 256;              // 16 rows x 64 steps
        const int r = li >> 6, c = li & 63;
        xs[li] = x[(b0 + r) * TLEN + t + c];
      }
      if (t + CHUNK < TLEN)
        __builtin_prefetch(&x[(b0 + (tid >> 4)) * TLEN + t + CHUNK + ((tid & 15) << 2)], 0, 0);
      __syncthreads();
    }
    const int tc = t & (CHUNK - 1);

    // ---- acc = xproj (gathered from LDS LUT in C-fragment layout) ----
    #pragma unroll
    for (int r = 0; r < 8; ++r) {
      const int row = r + 8 * lhalf;               // C layout: row = VGPR + 8*(lane/16)
      const int idx = xs[row * CHUNK + tc];
      acc[r] = lut[idx * HIDDEN + nbase + lm];
    }

    // ---- acc += h_{t-1} @ W_hh^T (this wave's 16 columns) ----
    #pragma unroll
    for (int kb = 0; kb < 4; ++kb)
      acc = __builtin_amdgcn_wmma_f32_16x16x32_f16(false, a[kb], false, bhh[kb],
                                                   (short)0, acc, false, false);

    // ---- h_t = tanh(acc); publish to ping-pong LDS buffer as f16 ----
    #pragma unroll
    for (int r = 0; r < 8; ++r) acc[r] = fast_tanh(acc[r]);
    _Float16* hw = hbuf[t & 1];
    #pragma unroll
    for (int r = 0; r < 8; ++r)
      hw[(r + 8 * lhalf) * HPAD + nbase + lm] = (_Float16)acc[r];
    __syncthreads();   // single barrier per step (double buffer covers WAR on hbuf)

    // ---- Reload full h_t as A-fragments (16-bit A 16x32 layout) ----
    const _Float16* hr = hbuf[t & 1];
    #pragma unroll
    for (int kb = 0; kb < 4; ++kb) {
      #pragma unroll
      for (int v = 0; v < 8; ++v) {
        const int k = kb * 32 + 16 * (v >> 2) + ((v & 3) << 1) + 8 * lhalf;
        const v2h p = *(const v2h*)&hr[lm * HPAD + k];
        a[kb][2 * v]     = p[0];
        a[kb][2 * v + 1] = p[1];
      }
    }

    // ---- Fused output head: out[:,t,:] = h_t @ W_fc^T + b_fc (waves 0,1) ----
    if (wave < 2) {
      v8f hc;
      #pragma unroll
      for (int r = 0; r < 8; ++r) hc[r] = bias;
      #pragma unroll
      for (int kb = 0; kb < 4; ++kb)
        hc = __builtin_amdgcn_wmma_f32_16x16x32_f16(false, a[kb], false, bfc[kb],
                                                    (short)0, hc, false, false);
      if (cfc < VOCAB) {
        #pragma unroll
        for (int r = 0; r < 8; ++r)
          out[(size_t)(b0 + r + 8 * lhalf) * outRowStride + (size_t)t * VOCAB + cfc] = hc[r];
      }
    }
  }

  // ---- Final hidden state (f32 accumulators still hold tanh of step T-1) ----
  #pragma unroll
  for (int r = 0; r < 8; ++r)
    hT[(b0 + r + 8 * lhalf) * HIDDEN + nbase + lm] = acc[r];
}

extern "C" void kernel_launch(void* const* d_in, const int* in_sizes, int n_in,
                              void* d_out, int out_size, void* d_ws, size_t ws_size,
                              hipStream_t stream) {
  (void)in_sizes; (void)n_in; (void)d_ws; (void)ws_size; (void)out_size;
  const int*   x     = (const int*)  d_in[0];
  const float* embed = (const float*)d_in[1];
  const float* W_ih  = (const float*)d_in[2];
  const float* W_hh  = (const float*)d_in[3];
  const float* b_ih  = (const float*)d_in[4];
  const float* b_hh  = (const float*)d_in[5];
  const float* W_fc  = (const float*)d_in[6];
  const float* b_fc  = (const float*)d_in[7];
  float* out = (float*)d_out;
  float* hT  = out + (size_t)256 * TLEN * VOCAB;   // tuple: (out, hT) concatenated

  rnn_fused<<<16, 256, 0, stream>>>(x, embed, W_ih, W_hh, b_ih, b_hh, W_fc, b_fc, out, hT);
}
